// Kenn_across_29661044146692
// MI455X (gfx1250) — matrix operand order repaired and checked by reading, not compiled, not run on hardware
//
#include <hip/hip_runtime.h>
#include <hip/hip_bf16.h>

// ---------------- CDNA5 WMMA types ----------------
typedef __attribute__((ext_vector_type(16))) __bf16 v16bf;
typedef __attribute__((ext_vector_type(8)))  float  v8f;

union V16BF {
    v16bf v;
    uint4 q[2];
};

// Round-to-nearest-even f32 -> bf16 (bit pattern in ushort)
__device__ __forceinline__ unsigned short f32_to_bf16(float f) {
    unsigned int u = __float_as_uint(f);
    u += 0x7FFFu + ((u >> 16) & 1u);
    return (unsigned short)(u >> 16);
}

// Problem constants (reference fixes D = H = 1024)
#define KD 1024           // D (GEMM1 K dimension)
#define KH 1024           // H (hidden width)
#define TM 128            // feature rows per workgroup
#define LDS_STRIDE (KD + 8)   // bf16 elements per LDS row (bank spread)

// ---------------------------------------------------------------------------
// Kernel 1: W1 [D][H] f32 (row-major)  ->  W1T [H][D] bf16 (row-major)
// ---------------------------------------------------------------------------
__global__ __launch_bounds__(256)
void k_transpose_w1(const float* __restrict__ w1, unsigned short* __restrict__ w1t) {
    __shared__ float tile[32][33];
    const int bh = blockIdx.x * 32;   // h tile
    const int bd = blockIdx.y * 32;   // d tile
    const int tx = threadIdx.x;       // 0..31
    const int ty = threadIdx.y;       // 0..7
#pragma unroll
    for (int i = ty; i < 32; i += 8)
        tile[i][tx] = w1[(size_t)(bd + i) * KH + (bh + tx)];
    __syncthreads();
#pragma unroll
    for (int i = ty; i < 32; i += 8)
        w1t[(size_t)(bh + i) * KD + (bd + tx)] = f32_to_bf16(tile[tx][i]);
}

// ---------------------------------------------------------------------------
// Kernel 2: fused  across = relu(features @ W1 + b1) @ W2 + b2
// written into pre[n_within + row].  bf16 WMMA, f32 accumulate.
// Each wave owns a 16-row tile and processes TWO 16-col H tiles per pass,
// with software-pipelined (double-buffered) A/B fragment loads.
// ---------------------------------------------------------------------------
__global__ __launch_bounds__(256)
void k_fused_mlp(const float* __restrict__ feat,
                 const unsigned short* __restrict__ w1t,   // [H][D] bf16
                 const float* __restrict__ b1,
                 const float* __restrict__ w2,             // [H][3]
                 const float* __restrict__ b2,
                 float* __restrict__ pre,                  // [n_total][3]
                 int n_across, int n_within) {
    extern __shared__ unsigned short s_feat[];   // TM x LDS_STRIDE bf16

    const int tid  = threadIdx.x;
    const int row0 = blockIdx.x * TM;

    // ---- stage 128 rows of features into LDS as bf16 ----
    for (int i = tid; i < TM * (KD / 4); i += 256) {
        const int row  = i >> 8;          // KD/4 == 256
        const int col4 = i & 255;
        float4 v = make_float4(0.f, 0.f, 0.f, 0.f);
        const int grow = row0 + row;
        if (grow < n_across)
            v = *(const float4*)(feat + (size_t)grow * KD + col4 * 4);
        uint2 p;
        p.x = (unsigned)f32_to_bf16(v.x) | ((unsigned)f32_to_bf16(v.y) << 16);
        p.y = (unsigned)f32_to_bf16(v.z) | ((unsigned)f32_to_bf16(v.w) << 16);
        *(uint2*)(s_feat + (size_t)row * LDS_STRIDE + col4 * 4) = p;
    }
    __syncthreads();

    const int wave = tid >> 5;            // 0..7  (wave32)
    const int lane = tid & 31;
    const int nrow = lane & 15;           // A: M index / B,C: N index for this lane
    const int ka   = (lane >> 4) << 3;    // A K-offset: 0 or 8 (split-K layout)
    const int kb16 = (lane >> 4) << 3;    // B K-offset: 0 or 8 (split-K layout)
    const int wave_row = wave * 16;       // 16-row tile per wave

    const unsigned short* aRow = s_feat + (size_t)(wave_row + nrow) * LDS_STRIDE;

    float acc[8][3];
#pragma unroll
    for (int r = 0; r < 8; ++r) { acc[r][0] = 0.f; acc[r][1] = 0.f; acc[r][2] = 0.f; }

    for (int ht = 0; ht < KH / 32; ++ht) {        // 2 H-tiles per iteration
        const int h0 = ht * 32;
        const int ha = h0 + nrow;                 // column owned by lane, tile 0
        const int hb = ha + 16;                   // column owned by lane, tile 1
        const unsigned short* bRow0 = w1t + (size_t)ha * KD;
        const unsigned short* bRow1 = w1t + (size_t)hb * KD;
        if (ht + 1 < KH / 32) {                   // prefetch next B tile pair
            __builtin_prefetch(bRow0 + (size_t)32 * KD, 0, 3);
            __builtin_prefetch(bRow1 + (size_t)32 * KD, 0, 3);
        }

        // epilogue operands in flight during the whole K loop
        const float bb0 = b1[ha],        bb1 = b1[hb];
        const float w20a = w2[ha * 3 + 0], w21a = w2[ha * 3 + 1], w22a = w2[ha * 3 + 2];
        const float w20b = w2[hb * 3 + 0], w21b = w2[hb * 3 + 1], w22b = w2[hb * 3 + 2];

        v8f c0 = {}, c1 = {};

        // ---- software-pipelined K loop ----
        V16BF a, b0, b1f;
        a.q[0]   = *(const uint4*)(aRow + ka);
        a.q[1]   = *(const uint4*)(aRow + ka + 16);
        b0.q[0]  = *(const uint4*)(bRow0 + kb16);
        b0.q[1]  = *(const uint4*)(bRow0 + kb16 + 16);
        b1f.q[0] = *(const uint4*)(bRow1 + kb16);
        b1f.q[1] = *(const uint4*)(bRow1 + kb16 + 16);

#pragma unroll 2
        for (int k = 32; k < KD; k += 32) {
            V16BF an, bn0, bn1;
            // issue next-step loads before consuming current fragments
            an.q[0]  = *(const uint4*)(aRow + k + ka);
            an.q[1]  = *(const uint4*)(aRow + k + ka + 16);
            bn0.q[0] = *(const uint4*)(bRow0 + k + kb16);
            bn0.q[1] = *(const uint4*)(bRow0 + k + kb16 + 16);
            bn1.q[0] = *(const uint4*)(bRow1 + k + kb16);
            bn1.q[1] = *(const uint4*)(bRow1 + k + kb16 + 16);

            c0 = __builtin_amdgcn_wmma_f32_16x16x32_bf16(
                     false, a.v, false, b0.v, (short)0, c0, false, false);
            c1 = __builtin_amdgcn_wmma_f32_16x16x32_bf16(
                     false, a.v, false, b1f.v, (short)0, c1, false, false);

            a = an; b0 = bn0; b1f = bn1;
        }
        c0 = __builtin_amdgcn_wmma_f32_16x16x32_bf16(
                 false, a.v, false, b0.v, (short)0, c0, false, false);
        c1 = __builtin_amdgcn_wmma_f32_16x16x32_bf16(
                 false, a.v, false, b1f.v, (short)0, c1, false, false);

        // epilogue: bias + ReLU + second GEMM (H x 3) accumulate, both tiles
#pragma unroll
        for (int r = 0; r < 8; ++r) {
            float hv0 = fmaxf(c0[r] + bb0, 0.f);
            float hv1 = fmaxf(c1[r] + bb1, 0.f);
            acc[r][0] = fmaf(hv0, w20a, fmaf(hv1, w20b, acc[r][0]));
            acc[r][1] = fmaf(hv0, w21a, fmaf(hv1, w21b, acc[r][1]));
            acc[r][2] = fmaf(hv0, w22a, fmaf(hv1, w22b, acc[r][2]));
        }
    }

    // reduce over the 16 lanes of each half (each lane held one h column)
#pragma unroll
    for (int r = 0; r < 8; ++r)
#pragma unroll
        for (int cc = 0; cc < 3; ++cc) {
            float v = acc[r][cc];
            v += __shfl_xor(v, 1, 32);
            v += __shfl_xor(v, 2, 32);
            v += __shfl_xor(v, 4, 32);
            v += __shfl_xor(v, 8, 32);
            acc[r][cc] = v;
        }

    if ((lane & 15) == 0) {
        // lane 0 -> rows wave_row+0..7 ; lane 16 -> rows wave_row+8..15
        const int rbase = row0 + wave_row + ((lane >> 4) << 3);
        const float c0b = b2[0], c1b = b2[1], c2b = b2[2];
#pragma unroll
        for (int r = 0; r < 8; ++r) {
            const int grow = rbase + r;
            if (grow < n_across) {
                float* dst = pre + (size_t)(n_within + grow) * 3;
                dst[0] = acc[r][0] + c0b;
                dst[1] = acc[r][1] + c1b;
                dst[2] = acc[r][2] + c2b;
            }
        }
    }
}

// ---------------------------------------------------------------------------
// Kernel 3: pre[0:n_within] = within_pre[:, :3]
// ---------------------------------------------------------------------------
__global__ __launch_bounds__(256)
void k_copy_within(const float* __restrict__ within, float* __restrict__ pre, int n_within) {
    int i = blockIdx.x * blockDim.x + threadIdx.x;
    if (i < n_within * 3) {
        int row = i / 3, col = i - row * 3;
        pre[i] = within[(size_t)row * 4 + col];
    }
}

// ---------------------------------------------------------------------------
// Kernel 4a: 3 KENN layers, all-register; emit final xz columns to temp
// ---------------------------------------------------------------------------
__global__ __launch_bounds__(256)
void k_kenn(const float* __restrict__ pre,
            const int* __restrict__ ixy, const int* __restrict__ iyz,
            const int* __restrict__ ixz, const float* __restrict__ cw,
            float* __restrict__ uxz, int M) {
    int m = blockIdx.x * blockDim.x + threadIdx.x;
    if (m >= M) return;
    const int j0 = ixy[m], j1 = iyz[m], j2 = ixz[m];
    float u[9];
    u[0] = pre[(size_t)j0 * 3 + 0]; u[1] = pre[(size_t)j0 * 3 + 1]; u[2] = pre[(size_t)j0 * 3 + 2];
    u[3] = pre[(size_t)j1 * 3 + 0]; u[4] = pre[(size_t)j1 * 3 + 1]; u[5] = pre[(size_t)j1 * 3 + 2];
    u[6] = pre[(size_t)j2 * 3 + 0]; u[7] = pre[(size_t)j2 * 3 + 1]; u[8] = pre[(size_t)j2 * 3 + 2];
#pragma unroll
    for (int l = 0; l < 3; ++l) {
        float d[9];
#pragma unroll
        for (int c = 0; c < 3; ++c) {
            // literals: sign * z for clause columns [c, c+3, c+6], signs = [-1,-1,+1]
            float s0 = -u[c], s1 = -u[c + 3], s2 = u[c + 6];
            float mx = fmaxf(s0, fmaxf(s1, s2));
            float e0 = expf(s0 - mx), e1 = expf(s1 - mx), e2 = expf(s2 - mx);
            float inv = 1.f / (e0 + e1 + e2);
            float w = cw[l * 3 + c];
            d[c]     = -w * e0 * inv;
            d[c + 3] = -w * e1 * inv;
            d[c + 6] =  w * e2 * inv;
        }
#pragma unroll
        for (int k = 0; k < 9; ++k) u[k] += d[k];
    }
    uxz[(size_t)m * 3 + 0] = u[6];
    uxz[(size_t)m * 3 + 1] = u[7];
    uxz[(size_t)m * 3 + 2] = u[8];
}

// ---------------------------------------------------------------------------
// Kernel 4b: scatter  pre[index_xz[m]] = uxz[m]
// ---------------------------------------------------------------------------
__global__ __launch_bounds__(256)
void k_scatter(const int* __restrict__ ixz, const float* __restrict__ uxz,
               float* __restrict__ pre, int M) {
    int m = blockIdx.x * blockDim.x + threadIdx.x;
    if (m >= M) return;
    const int j = ixz[m];
    pre[(size_t)j * 3 + 0] = uxz[(size_t)m * 3 + 0];
    pre[(size_t)j * 3 + 1] = uxz[(size_t)m * 3 + 1];
    pre[(size_t)j * 3 + 2] = uxz[(size_t)m * 3 + 2];
}

// ---------------------------------------------------------------------------
// Kernel 5: out = pre[n_within:], softmax(out) appended
// ---------------------------------------------------------------------------
__global__ __launch_bounds__(256)
void k_finalize(const float* __restrict__ pre, float* __restrict__ out,
                int n_across, int n_within) {
    int i = blockIdx.x * blockDim.x + threadIdx.x;
    if (i >= n_across) return;
    const float* row = pre + (size_t)(n_within + i) * 3;
    float a = row[0], b = row[1], c = row[2];
    out[(size_t)i * 3 + 0] = a;
    out[(size_t)i * 3 + 1] = b;
    out[(size_t)i * 3 + 2] = c;
    float mx = fmaxf(a, fmaxf(b, c));
    float e0 = expf(a - mx), e1 = expf(b - mx), e2 = expf(c - mx);
    float inv = 1.f / (e0 + e1 + e2);
    float* sm = out + (size_t)n_across * 3;
    sm[(size_t)i * 3 + 0] = e0 * inv;
    sm[(size_t)i * 3 + 1] = e1 * inv;
    sm[(size_t)i * 3 + 2] = e2 * inv;
}

// ---------------------------------------------------------------------------
extern "C" void kernel_launch(void* const* d_in, const int* in_sizes, int n_in,
                              void* d_out, int out_size, void* d_ws, size_t ws_size,
                              hipStream_t stream) {
    const float* feat   = (const float*)d_in[0];
    const float* within = (const float*)d_in[1];
    const int*   ixy    = (const int*)d_in[2];
    const int*   iyz    = (const int*)d_in[3];
    const int*   ixz    = (const int*)d_in[4];
    const float* w1     = (const float*)d_in[5];
    const float* b1     = (const float*)d_in[6];
    const float* w2     = (const float*)d_in[7];
    const float* b2     = (const float*)d_in[8];
    const float* cw     = (const float*)d_in[9];

    const int H        = in_sizes[6];            // 1024
    const int D        = in_sizes[5] / H;        // 1024
    const int n_across = in_sizes[0] / D;        // 200000
    const int n_within = in_sizes[1] / 4;        // 100000
    const int M        = in_sizes[2];            // 262144
    (void)H; (void)D;

    // workspace carve-up
    char* ws = (char*)d_ws;
    unsigned short* w1t = (unsigned short*)ws;                 // [H][D] bf16
    size_t off = (size_t)KD * KH * sizeof(unsigned short);
    float* pre = (float*)(ws + off);                           // [n_total][3] f32
    off += (size_t)(n_within + n_across) * 3 * sizeof(float);
    off = (off + 255) & ~(size_t)255;
    float* uxz = (float*)(ws + off);                           // [M][3] f32

    float* out = (float*)d_out;

    // 1. transpose + convert W1 -> bf16 [H][D]
    k_transpose_w1<<<dim3(KH / 32, KD / 32), dim3(32, 8), 0, stream>>>(w1, w1t);

    // 2. within_pre[:, :3] -> pre[0:n_within]
    k_copy_within<<<(n_within * 3 + 255) / 256, 256, 0, stream>>>(within, pre, n_within);

    // 3. fused bf16-WMMA MLP -> pre[n_within:]
    const size_t lds_bytes = (size_t)TM * LDS_STRIDE * sizeof(unsigned short); // ~258 KB
    k_fused_mlp<<<(n_across + TM - 1) / TM, 256, lds_bytes, stream>>>(
        feat, w1t, b1, w2, b2, pre, n_across, n_within);

    // 4. KENN layers (gather phase) then scatter phase
    k_kenn<<<(M + 255) / 256, 256, 0, stream>>>(pre, ixy, iyz, ixz, cw, uxz, M);
    k_scatter<<<(M + 255) / 256, 256, 0, stream>>>(ixz, uxz, pre, M);

    // 5. outputs: raw + softmax
    k_finalize<<<(n_across + 255) / 256, 256, 0, stream>>>(pre, out, n_across, n_within);
}
